// MixtureOfExperts_62311385530890
// MI455X (gfx1250) — compile-verified, matrix operand order checked by reading, not run on hardware
//
#include <hip/hip_runtime.h>
#include <hip/hip_bf16.h>

// ---------------------------------------------------------------------------
// MoE forward for MI455X (gfx1250, wave32, WMMA).
// Pipeline per call (all on `stream`):
//   1. zero d_out, zero per-expert counters
//   2. convert x -> bf16 (ws)
//   3. transpose+convert w1 -> w1t[H][C] bf16, w2 -> w2t[C][H] bf16 (ws)
//   4. gating: one wave per token, top-2 softmax, build per-expert token lists
//   5. fused expert kernel: gather 64-token tile, GEMM1(+bias,relu) -> LDS,
//      GEMM2 accumulated in registers over H-chunks, scaled atomic scatter.
// ---------------------------------------------------------------------------

typedef unsigned int u32x4 __attribute__((ext_vector_type(4)));
typedef unsigned int u32x2 __attribute__((ext_vector_type(2)));
typedef float        f32x4 __attribute__((ext_vector_type(4)));
typedef float        v8f   __attribute__((ext_vector_type(8)));
typedef __bf16       v16bf __attribute__((ext_vector_type(16)));

#define MT  64            // tokens per block in expert kernel
#define HC  512           // hidden-dim chunk held in LDS
#define LDX (1024 + 8)    // padded LDS stride for x tile (bf16 elems)
#define LDH (HC + 8)      // padded LDS stride for h chunk (bf16 elems)

static constexpr int Bb  = 4;
static constexpr int Tt  = 4096;
static constexpr int Cd  = 1024;
static constexpr int Ee  = 8;
static constexpr int Hd  = 4096;
static constexpr int Nn  = Bb * Tt;   // 16384 tokens

__device__ __forceinline__ unsigned short cvt_bf16(float f) {
  unsigned u = __float_as_uint(f);
  unsigned r = u + 0x7FFFu + ((u >> 16) & 1u);   // round-to-nearest-even
  return (unsigned short)(r >> 16);
}

union Frag { u32x4 q[2]; v16bf v; };

// Build a 16x32-bf16 A/B WMMA operand: per-lane two contiguous 16B chunks.
// K sub-block pattern per CDNA5 ISA: lanes 0-15 -> K {0..7,16..23},
// lanes 16-31 -> K {8..15,24..31} (off0/off1 carry the half-wave shift).
__device__ __forceinline__ v16bf load_frag(const unsigned short* row,
                                           int off0, int off1) {
  Frag f;
  f.q[0] = *(const u32x4*)(row + off0);
  f.q[1] = *(const u32x4*)(row + off1);
  return f.v;
}

__device__ __forceinline__ v8f wmma_bf16(v16bf a, v16bf b, v8f c) {
  return __builtin_amdgcn_wmma_f32_16x16x32_bf16(
      /*neg_a=*/false, a, /*neg_b=*/false, b,
      /*c_mod=*/(short)0, c, /*reuse_a=*/false, /*reuse_b=*/false);
}

// ---------------------------------------------------------------------------
__global__ void zero_f32_kernel(float* __restrict__ p, size_t n) {
  size_t stride = (size_t)gridDim.x * blockDim.x;
  for (size_t i = blockIdx.x * (size_t)blockDim.x + threadIdx.x; i < n; i += stride)
    p[i] = 0.0f;
}

__global__ void zero_i32_kernel(int* __restrict__ p, int n) {
  int i = blockIdx.x * blockDim.x + threadIdx.x;
  if (i < n) p[i] = 0;
}

// x (f32, row-major [N][C]) -> xbf (bf16 bits), 4 elems per thread
__global__ void cvt_x_kernel(const float* __restrict__ x,
                             unsigned short* __restrict__ xbf, size_t n4) {
  size_t i = blockIdx.x * (size_t)blockDim.x + threadIdx.x;
  if (i >= n4) return;
  f32x4 v = *(const f32x4*)(x + 4 * i);
  u32x2 p;
  p[0] = (unsigned)cvt_bf16(v[0]) | ((unsigned)cvt_bf16(v[1]) << 16);
  p[1] = (unsigned)cvt_bf16(v[2]) | ((unsigned)cvt_bf16(v[3]) << 16);
  *(u32x2*)(xbf + 4 * i) = p;
}

// in[e][R][Cc] f32 -> out[e][Cc][R] bf16  (32x32 LDS tile transpose)
__global__ void transpose_cvt_kernel(const float* __restrict__ in,
                                     unsigned short* __restrict__ out,
                                     int R, int Cc) {
  __shared__ float t[32][33];
  int e  = blockIdx.z;
  int c0 = blockIdx.x * 32;
  int r0 = blockIdx.y * 32;
  const float*    ip = in  + (size_t)e * R * Cc;
  unsigned short* op = out + (size_t)e * Cc * R;
  int tx = threadIdx.x, ty = threadIdx.y;
#pragma unroll
  for (int k = 0; k < 4; ++k)
    t[ty + 8 * k][tx] = ip[(size_t)(r0 + ty + 8 * k) * Cc + c0 + tx];
  __syncthreads();
#pragma unroll
  for (int k = 0; k < 4; ++k)
    op[(size_t)(c0 + ty + 8 * k) * R + r0 + tx] = cvt_bf16(t[tx][ty + 8 * k]);
}

// One wave32 per token: logits, top-2 softmax, append to per-expert lists.
__global__ void gate_kernel(const float* __restrict__ x,
                            const float* __restrict__ gw,
                            const float* __restrict__ gb,
                            int*   __restrict__ counts,
                            int*   __restrict__ lists,
                            float* __restrict__ wts) {
  int tok  = (int)((blockIdx.x * (size_t)blockDim.x + threadIdx.x) >> 5);
  int lane = threadIdx.x & 31;
  if (tok >= Nn) return;

  const float* xr = x + (size_t)tok * Cd;
  float acc[8];
#pragma unroll
  for (int e = 0; e < 8; ++e) acc[e] = 0.0f;

  for (int k = lane; k < Cd; k += 32) {
    float xv = xr[k];
    const float* g = gw + (size_t)k * Ee;
    f32x4 g0 = *(const f32x4*)g;
    f32x4 g1 = *(const f32x4*)(g + 4);
    acc[0] += xv * g0[0]; acc[1] += xv * g0[1];
    acc[2] += xv * g0[2]; acc[3] += xv * g0[3];
    acc[4] += xv * g1[0]; acc[5] += xv * g1[1];
    acc[6] += xv * g1[2]; acc[7] += xv * g1[3];
  }
#pragma unroll
  for (int e = 0; e < 8; ++e) {
    float v = acc[e];
#pragma unroll
    for (int off = 16; off; off >>= 1) v += __shfl_xor(v, off, 32);
    acc[e] = v;
  }

  if (lane == 0) {
#pragma unroll
    for (int e = 0; e < 8; ++e) acc[e] += gb[e];
    int i0 = 0;
#pragma unroll
    for (int e = 1; e < 8; ++e) if (acc[e] > acc[i0]) i0 = e;
    int i1 = (i0 == 0) ? 1 : 0;
#pragma unroll
    for (int e = 0; e < 8; ++e)
      if (e != i0 && acc[e] > acc[i1]) i1 = e;
    float d  = __expf(acc[i1] - acc[i0]);   // <= 1
    float w0 = 1.0f / (1.0f + d);
    float w1 = 1.0f - w0;
    int s0 = atomicAdd(counts + i0, 1);
    lists[(size_t)i0 * Nn + s0] = tok;
    wts  [(size_t)i0 * Nn + s0] = w0;
    int s1 = atomicAdd(counts + i1, 1);
    lists[(size_t)i1 * Nn + s1] = tok;
    wts  [(size_t)i1 * Nn + s1] = w1;
  }
}

// ---------------------------------------------------------------------------
// Fused per-expert MLP: 64-token tile, H chunked by 512 through LDS.
// block = 512 threads (16 waves); grid = (N/MT tiles, E experts), early exit.
// ---------------------------------------------------------------------------
__global__ __launch_bounds__(512, 1)
void moe_expert_kernel(const unsigned short* __restrict__ xbf,
                       const unsigned short* __restrict__ w1t,  // [E][H][C]
                       const float*          __restrict__ b1,   // [E][H]
                       const unsigned short* __restrict__ w2t,  // [E][C][H]
                       const float*          __restrict__ b2,   // [E][C]
                       const int*   __restrict__ counts,
                       const int*   __restrict__ lists,
                       const float* __restrict__ wts,
                       float* __restrict__ out) {
  const int e    = blockIdx.y;
  const int cnt  = counts[e];
  const int tile = blockIdx.x;
  if (tile * MT >= cnt) return;

  extern __shared__ unsigned short smem[];
  unsigned short* xs  = smem;                 // MT * LDX bf16
  unsigned short* hs  = xs + MT * LDX;        // MT * LDH bf16
  int*            tok = (int*)(hs + MT * LDH);
  float*          scl = (float*)(tok + MT);

  const int tid = threadIdx.x;
  if (tid < MT) {
    int slot = tile * MT + tid;
    int s = slot < cnt ? slot : cnt - 1;
    tok[tid] = lists[(size_t)e * Nn + s];
    scl[tid] = slot < cnt ? wts[(size_t)e * Nn + s] : 0.0f;
  }
  __syncthreads();

  // gather x rows for this tile into LDS (8 bytes / thread / iter, coalesced)
  {
    const int QW = MT * Cd / 4;           // 4-bf16 groups
    for (int q = tid; q < QW; q += 512) {
      int row = q >> 8;                   // Cd/4 = 256 groups per row
      int c4  = (q & 255) << 2;
      *(u32x2*)(xs + row * LDX + c4) =
          *(const u32x2*)(xbf + (size_t)tok[row] * Cd + c4);
    }
  }
  __syncthreads();

  const int wave  = tid >> 5;
  const int lane  = tid & 31;
  const int half  = lane >> 4;
  const int r     = lane & 15;
  const int koff0 = half * 8;
  const int koff1 = 16 + half * 8;
  const int nb    = wave * 64;            // this wave's 64 output columns

  // persistent GEMM2 accumulators, seeded with b2 (bias applied once/expert)
  v8f oacc[4][4];
#pragma unroll
  for (int nt = 0; nt < 4; ++nt) {
    float bv = b2[(size_t)e * Cd + nb + nt * 16 + r];
#pragma unroll
    for (int mt = 0; mt < 4; ++mt)
#pragma unroll
      for (int i = 0; i < 8; ++i) oacc[mt][nt][i] = bv;
  }

  for (int hc = 0; hc < Hd; hc += HC) {
    // ---- GEMM1: this wave produces h columns [hc+wave*32, +32) ----
    v8f acc1[4][2];
#pragma unroll
    for (int nt = 0; nt < 2; ++nt) {
      float bv = b1[(size_t)e * Hd + hc + wave * 32 + nt * 16 + r];
#pragma unroll
      for (int mt = 0; mt < 4; ++mt)
#pragma unroll
        for (int i = 0; i < 8; ++i) acc1[mt][nt][i] = bv;
    }
    for (int kk = 0; kk < Cd; kk += 32) {
      v16bf bfr[2];
#pragma unroll
      for (int nt = 0; nt < 2; ++nt) {
        int nrow = hc + wave * 32 + nt * 16 + r;
        bfr[nt] = load_frag(w1t + ((size_t)e * Hd + nrow) * Cd + kk,
                            koff0, koff1);
      }
#pragma unroll
      for (int mt = 0; mt < 4; ++mt) {
        v16bf afr = load_frag(xs + (mt * 16 + r) * LDX + kk, koff0, koff1);
#pragma unroll
        for (int nt = 0; nt < 2; ++nt)
          acc1[mt][nt] = wmma_bf16(afr, bfr[nt], acc1[mt][nt]);
      }
    }
    // relu + bf16 store of the h chunk into LDS
#pragma unroll
    for (int mt = 0; mt < 4; ++mt)
#pragma unroll
      for (int nt = 0; nt < 2; ++nt)
#pragma unroll
        for (int i = 0; i < 8; ++i) {
          int m = mt * 16 + half * 8 + i;
          int n = wave * 32 + nt * 16 + r;
          float v = acc1[mt][nt][i];
          hs[m * LDH + n] = cvt_bf16(v > 0.0f ? v : 0.0f);
        }
    __syncthreads();

    // ---- GEMM2 partial over this H chunk ----
    for (int k2 = 0; k2 < HC; k2 += 32) {
      v16bf bfr2[4];
#pragma unroll
      for (int nt = 0; nt < 4; ++nt) {
        int n = nb + nt * 16 + r;
        bfr2[nt] = load_frag(w2t + ((size_t)e * Cd + n) * Hd + hc + k2,
                             koff0, koff1);
      }
#pragma unroll
      for (int mt = 0; mt < 4; ++mt) {
        v16bf afr = load_frag(hs + (mt * 16 + r) * LDH + k2, koff0, koff1);
#pragma unroll
        for (int nt = 0; nt < 4; ++nt)
          oacc[mt][nt] = wmma_bf16(afr, bfr2[nt], oacc[mt][nt]);
      }
    }
    __syncthreads();   // hs reused next chunk
  }

  // scaled scatter: each output element gets exactly 2 commutative f32 adds
#pragma unroll
  for (int mt = 0; mt < 4; ++mt) {
#pragma unroll
    for (int i = 0; i < 8; ++i) {
      int   m = mt * 16 + half * 8 + i;
      int   t = tok[m];
      float s = scl[m];
#pragma unroll
      for (int nt = 0; nt < 4; ++nt) {
        int n = nb + nt * 16 + r;
        atomicAdd(out + (size_t)t * Cd + n, s * oacc[mt][nt][i]);
      }
    }
  }
}

// ---------------------------------------------------------------------------
extern "C" void kernel_launch(void* const* d_in, const int* in_sizes, int n_in,
                              void* d_out, int out_size, void* d_ws, size_t ws_size,
                              hipStream_t stream) {
  const float* x  = (const float*)d_in[0];
  const float* gw = (const float*)d_in[1];
  const float* gb = (const float*)d_in[2];
  const float* w1 = (const float*)d_in[3];
  const float* b1 = (const float*)d_in[4];
  const float* w2 = (const float*)d_in[5];
  const float* b2 = (const float*)d_in[6];
  float* out = (float*)d_out;

  // workspace carve (≈161 MB)
  char* ws = (char*)d_ws;
  int* counts = (int*)ws;                       ws += 256;
  int* lists  = (int*)ws;                       ws += (size_t)Ee * Nn * 4;
  float* wts  = (float*)ws;                     ws += (size_t)Ee * Nn * 4;
  unsigned short* xbf = (unsigned short*)ws;    ws += (size_t)Nn * Cd * 2;
  unsigned short* w1t = (unsigned short*)ws;    ws += (size_t)Ee * Hd * Cd * 2;
  unsigned short* w2t = (unsigned short*)ws;    ws += (size_t)Ee * Cd * Hd * 2;

  zero_f32_kernel<<<2048, 256, 0, stream>>>(out, (size_t)Nn * Cd);
  zero_i32_kernel<<<1, 64, 0, stream>>>(counts, Ee);

  {
    size_t n4 = (size_t)Nn * Cd / 4;
    cvt_x_kernel<<<(unsigned)((n4 + 255) / 256), 256, 0, stream>>>(x, xbf, n4);
  }
  // w1 [E][C][H] -> w1t [E][H][C]
  transpose_cvt_kernel<<<dim3(Hd / 32, Cd / 32, Ee), dim3(32, 8), 0, stream>>>(
      w1, w1t, Cd, Hd);
  // w2 [E][H][C] -> w2t [E][C][H]
  transpose_cvt_kernel<<<dim3(Cd / 32, Hd / 32, Ee), dim3(32, 8), 0, stream>>>(
      w2, w2t, Hd, Cd);

  gate_kernel<<<(Nn * 32) / 256, 256, 0, stream>>>(x, gw, gb, counts, lists, wts);

  size_t shmem = (size_t)(MT * LDX + MT * LDH) * 2 + (size_t)MT * 8;
  moe_expert_kernel<<<dim3(Nn / MT, Ee), 512, shmem, stream>>>(
      xbf, w1t, b1, w2t, b2, counts, lists, wts, out);
}
